// MultiHeadAttention_35699768165245
// MI455X (gfx1250) — matrix operand order repaired and checked by reading, not compiled, not run on hardware
//
#include <hip/hip_runtime.h>
#include <hip/hip_bf16.h>

typedef __attribute__((ext_vector_type(16))) _Float16 v16h;
typedef __attribute__((ext_vector_type(8)))  _Float16 v8h;
typedef __attribute__((ext_vector_type(8)))  float    v8f;
typedef __attribute__((ext_vector_type(4)))  float    v4f;

#define TT 96
#define DD 64
#define HH 8
#define DKK 8

static __device__ __forceinline__ v8f wmma_f16(v16h a, v16h b, v8f c) {
  // D = A(16x32 f16) x B(32x16 f16) + C(16x16 f32)
  return __builtin_amdgcn_wmma_f32_16x16x32_f16(false, a, false, b, (short)0, c,
                                                false, false);
}

static __device__ __forceinline__ v16h cat8(v8h lo, v8h hi) {
  return __builtin_shufflevector(lo, hi, 0, 1, 2, 3, 4, 5, 6, 7, 8, 9, 10, 11,
                                 12, 13, 14, 15);
}

// A fragment 16x32 (MxK), all K valid, from row-major f16 LDS.
// ISA layout: element e -> K = k0 + ((e>>3)<<4) + (lane>>4)*8 + (e&7)
// => two contiguous 8-element runs per lane -> two ds_load_b128.
static __device__ __forceinline__ v16h load_a_full(const _Float16* base, int row0,
                                                   int ld, int k0, int lane) {
  int m = lane & 15, hi = lane >> 4;
  const _Float16* rp = base + (row0 + m) * ld + k0 + hi * 8;
  v8h lo = *(const v8h*)rp;
  v8h h8 = *(const v8h*)(rp + 16);
  return cat8(lo, h8);
}

// Fragment with only K=0..7 valid (dk=8 zero-padded to K=32). Works for both
// the A layout (K = hi*8 + e) and the B layout (K = hi*8 + e): in both cases
// valid elements are e=0..7 on lanes with hi==0, read from 8 contiguous f16.
static __device__ __forceinline__ v16h load_run8(const _Float16* rp, int hi) {
  v8h lo = *(const v8h*)(rp + hi * 8);  // hi==1 lanes: value discarded
  v16h r = {};
  if (hi == 0) {
#pragma unroll
    for (int e = 0; e < 8; ++e) r[e] = lo[e];
  }
  return r;
}

// B fragment 32x16 (KxN) read from an (N x K) row-major (i.e. transposed)
// f16 LDS buffer: element e -> K = k0 + ((e>>3)<<4) + (lane>>4)*8 + (e&7), N = col.
// => two contiguous 8-element runs per lane -> two ds_load_b128.
static __device__ __forceinline__ v16h load_bT_full(const _Float16* baseT, int col,
                                                    int ld, int k0, int lane) {
  const _Float16* rp = baseT + col * ld + k0 + (lane >> 4) * 8;
  v8h lo = *(const v8h*)rp;
  v8h h8 = *(const v8h*)(rp + 16);
  return cat8(lo, h8);
}

// A fragment from row-major f32 global buffer (ld = DD), cvt f32->f16.
static __device__ __forceinline__ v16h load_a_g32(const float* src, int row0,
                                                  int k0, int lane) {
  int m = lane & 15, hi = lane >> 4;
  const float* rp = src + (size_t)(row0 + m) * DD + k0 + hi * 8;
  v4f a0 = *(const v4f*)rp;
  v4f a1 = *(const v4f*)(rp + 4);
  v4f a2 = *(const v4f*)(rp + 16);
  v4f a3 = *(const v4f*)(rp + 20);
  v16h a;
#pragma unroll
  for (int e = 0; e < 4; ++e) {
    a[e]      = (_Float16)a0[e];
    a[e + 4]  = (_Float16)a1[e];
    a[e + 8]  = (_Float16)a2[e];
    a[e + 12] = (_Float16)a3[e];
  }
  return a;
}

// One 96x64 projection: dst = f16(src @ W + b). 24 tiles, 3 per wave.
// STORE_T=false: dst row-major (T x D). STORE_T=true: dst transposed (D x T),
// which lets the store be a single packed ds_store_b128 per lane.
template <bool STORE_T>
static __device__ __forceinline__ void proj96x64(const float* src,
                                                 const _Float16* Wt,
                                                 const float* bias, _Float16* dst,
                                                 int wave, int lane) {
  int nlo = lane & 15, hi = lane >> 4;
  for (int it = 0; it < 3; ++it) {
    int t = wave * 3 + it;
    int mi = t >> 2, nn = t & 3;
    v8f acc = {};
#pragma unroll
    for (int kk = 0; kk < 2; ++kk) {
      v16h a = load_a_g32(src, mi * 16, kk * 32, lane);
      v16h b = load_bT_full(Wt, nn * 16 + nlo, DD, kk * 32, lane);
      acc = wmma_f16(a, b, acc);
    }
    float bb = bias[nn * 16 + nlo];
    if (STORE_T) {
      v8h p;
#pragma unroll
      for (int r = 0; r < 8; ++r) p[r] = (_Float16)(acc[r] + bb);
      *(v8h*)(dst + (nn * 16 + nlo) * TT + mi * 16 + hi * 8) = p;
    } else {
#pragma unroll
      for (int r = 0; r < 8; ++r)
        dst[(mi * 16 + hi * 8 + r) * DD + nn * 16 + nlo] = (_Float16)(acc[r] + bb);
    }
  }
}

__global__ __launch_bounds__(256)
void MultiHeadAttention_gauss_wmma_kernel(
    const float* __restrict__ q_g, const float* __restrict__ k_g,
    const float* __restrict__ v_g,
    const float* __restrict__ Wq, const float* __restrict__ bq,
    const float* __restrict__ Wk, const float* __restrict__ bk,
    const float* __restrict__ Wv, const float* __restrict__ bv,
    const float* __restrict__ Wo, const float* __restrict__ bo,
    const float* __restrict__ Ws, const float* __restrict__ bs,
    float* __restrict__ out) {
  extern __shared__ char smem[];
  _Float16* sWqT = (_Float16*)smem;        // 64x64, transposed (N x K)
  _Float16* sWkT = sWqT + 4096;
  _Float16* sWvT = sWkT + 4096;
  _Float16* sWoT = sWvT + 4096;
  _Float16* sWsT = sWoT + 4096;            // 8x64, transposed
  _Float16* sQ   = sWsT + 512;             // 96x64 row-major
  _Float16* sK   = sQ + TT * DD;           // 96x64 row-major
  _Float16* sVt  = sK + TT * DD;           // 64x96 (transposed V)
  _Float16* sX   = sVt + DD * TT;          // 96x64 row-major
  _Float16* sP   = sX + TT * DD;           // 8 waves * 16x96
  float* sCoef = (float*)(sP + 8 * 16 * TT);  // 96*8
  float* sInv  = sCoef + TT * HH;             // 96*8
  float* sBias = sInv + TT * HH;              // 4*64 (bq,bk,bv,bo)

  const int bn   = blockIdx.x;
  const int lane = threadIdx.x & 31;
  const int wave = threadIdx.x >> 5;
  const int nlo  = lane & 15;
  const int hi   = lane >> 4;
  const size_t seq_off = (size_t)bn * TT * DD;

  // ---- Stage weights transposed (f32 -> f16) + biases ----
  for (int idx = threadIdx.x; idx < DD * DD; idx += 256) {
    int kq = idx >> 6, nq = idx & 63;          // W[k][n]
    sWqT[nq * DD + kq] = (_Float16)Wq[idx];
    sWkT[nq * DD + kq] = (_Float16)Wk[idx];
    sWvT[nq * DD + kq] = (_Float16)Wv[idx];
    sWoT[nq * DD + kq] = (_Float16)Wo[idx];
  }
  for (int idx = threadIdx.x; idx < DD * HH; idx += 256) {
    int kq = idx >> 3, nq = idx & 7;           // Ws[k][h]
    sWsT[nq * DD + kq] = (_Float16)Ws[idx];
  }
  if (threadIdx.x < 64) {
    sBias[0 * 64 + threadIdx.x] = bq[threadIdx.x];
    sBias[1 * 64 + threadIdx.x] = bk[threadIdx.x];
    sBias[2 * 64 + threadIdx.x] = bv[threadIdx.x];
    sBias[3 * 64 + threadIdx.x] = bo[threadIdx.x];
  }
  __syncthreads();

  // ---- QKV projections on the matrix pipe (V stored transposed) ----
  proj96x64<false>(q_g + seq_off, sWqT, sBias + 0,   sQ,  wave, lane);
  proj96x64<false>(k_g + seq_off, sWkT, sBias + 64,  sK,  wave, lane);
  proj96x64<true >(v_g + seq_off, sWvT, sBias + 128, sVt, wave, lane);

  // ---- sigma head via WMMA: (96x64) @ WsT-padded; waves 0..5, 1 tile each ----
  if (wave < 6) {
    int mi = wave;
    v8f acc = {};
#pragma unroll
    for (int kk = 0; kk < 2; ++kk) {
      v16h a = load_a_g32(q_g + seq_off, mi * 16, kk * 32, lane);
      v16h b = load_bT_full(sWsT, nlo & 7, DD, kk * 32, lane);  // cols 8..15 dup
      acc = wmma_f16(a, b, acc);
    }
    if (nlo < HH) {
      float bsv = bs[nlo];
#pragma unroll
      for (int r = 0; r < 8; ++r) {
        int row = mi * 16 + hi * 8 + r;
        float s   = acc[r] + bsv;
        float sg  = 1.0f / (1.0f + __expf(-5.0f * s)) + 1e-5f;  // sigmoid
        float sig = __expf(sg * 1.0986122886681098f) - 1.0f;    // 3^sg - 1
        sCoef[row * HH + nlo] = 0.3989422804014327f / sig;      // 1/(sqrt(2pi) sig)
        sInv[row * HH + nlo]  = 0.5f / (sig * sig);
      }
    }
  }
  __syncthreads();

  // ---- Attention: one head per wave (H == 8 waves) ----
  {
    const int h = wave;
    _Float16* pw = sP + wave * 16 * TT;          // private 16x96 P row-tile
    const float scale = 0.35355339059327373f;    // 1/sqrt(8)
    for (int i = 0; i < 6; ++i) {
      // Q row-tile fragment: dk=8 zero-padded into K=32
      v16h aq = load_run8(sQ + (i * 16 + nlo) * DD + h * DKK, hi);
      v8f st[6];
#pragma unroll
      for (int j = 0; j < 6; ++j) {
        v16h bk_ = load_run8(sK + (j * 16 + nlo) * DD + h * DKK, hi);
        v8f c = {};
        st[j] = wmma_f16(aq, bk_, c);
      }
      // gaussian bias + causal mask + scale
      float coef8[8], inv8[8];
#pragma unroll
      for (int r = 0; r < 8; ++r) {
        int row = i * 16 + hi * 8 + r;
        coef8[r] = sCoef[row * HH + h];
        inv8[r]  = sInv[row * HH + h];
      }
#pragma unroll
      for (int j = 0; j < 6; ++j) {
        int col = j * 16 + nlo;
        float c2 = (float)(col * col);
#pragma unroll
        for (int r = 0; r < 8; ++r) {
          int row = i * 16 + hi * 8 + r;
          float sc;
          if (col > row) sc = -__builtin_inff();
          else sc = (st[j][r] + coef8[r] * __expf(-c2 * inv8[r])) * scale;
          st[j][r] = sc;
        }
      }
      // row softmax: 96 cols across 6 tiles x 16-lane group (wave32 shuffles)
      float mx[8], sum[8];
#pragma unroll
      for (int r = 0; r < 8; ++r) {
        float m = st[0][r];
#pragma unroll
        for (int j = 1; j < 6; ++j) m = fmaxf(m, st[j][r]);
#pragma unroll
        for (int d = 8; d >= 1; d >>= 1) m = fmaxf(m, __shfl_xor(m, d, 32));
        mx[r] = m;
      }
#pragma unroll
      for (int r = 0; r < 8; ++r) {
        float s = 0.0f;
#pragma unroll
        for (int j = 0; j < 6; ++j) {
          float p = __expf(st[j][r] - mx[r]);
          st[j][r] = p;
          s += p;
        }
#pragma unroll
        for (int d = 8; d >= 1; d >>= 1) s += __shfl_xor(s, d, 32);
        sum[r] = s;
      }
#pragma unroll
      for (int r = 0; r < 8; ++r) {
        float inv = 1.0f / sum[r];
        int rr = hi * 8 + r;
#pragma unroll
        for (int j = 0; j < 6; ++j)
          pw[rr * TT + j * 16 + nlo] = (_Float16)(st[j][r] * inv);
      }
      // per-wave LDS RAW: P must land before the A-fragment reload
      asm volatile("s_wait_dscnt 0" ::: "memory");

      // P(16x96) @ V_h(96x8): 3 K32 steps; B from transposed V => b128 loads
      v8f acc = {};
#pragma unroll
      for (int kk = 0; kk < 3; ++kk) {
        v16h ap  = load_a_full(pw, 0, TT, kk * 32, lane);
        v16h bv_ = load_bT_full(sVt, h * DKK + (nlo & 7), TT, kk * 32, lane);
        acc = wmma_f16(ap, bv_, acc);
      }
      if (nlo < DKK) {
#pragma unroll
        for (int r = 0; r < 8; ++r) {
          int row = i * 16 + hi * 8 + r;
          sX[row * DD + h * DKK + nlo] = (_Float16)acc[r];
        }
      }
    }
  }
  __syncthreads();

  // ---- Output projection: x(96x64) @ Wo + bo -> f32 global ----
  for (int it = 0; it < 3; ++it) {
    int t = wave * 3 + it;
    int mi = t >> 2, nn = t & 3;
    v8f acc = {};
#pragma unroll
    for (int kk = 0; kk < 2; ++kk) {
      v16h a = load_a_full(sX, mi * 16, DD, kk * 32, lane);
      v16h b = load_bT_full(sWoT, nn * 16 + nlo, DD, kk * 32, lane);
      acc = wmma_f16(a, b, acc);
    }
    float bb = sBias[3 * 64 + nn * 16 + nlo];
#pragma unroll
    for (int r = 0; r < 8; ++r) {
      int row = mi * 16 + hi * 8 + r;
      out[seq_off + row * DD + nn * 16 + nlo] = acc[r] + bb;
    }
  }
}

extern "C" void kernel_launch(void* const* d_in, const int* in_sizes, int n_in,
                              void* d_out, int out_size, void* d_ws, size_t ws_size,
                              hipStream_t stream) {
  const float* q  = (const float*)d_in[0];
  const float* k  = (const float*)d_in[1];
  const float* v  = (const float*)d_in[2];
  const float* Wq = (const float*)d_in[3];
  const float* bq = (const float*)d_in[4];
  const float* Wk = (const float*)d_in[5];
  const float* bk = (const float*)d_in[6];
  const float* Wv = (const float*)d_in[7];
  const float* bv = (const float*)d_in[8];
  const float* Wo = (const float*)d_in[9];
  const float* bo = (const float*)d_in[10];
  const float* Ws = (const float*)d_in[11];
  const float* bs = (const float*)d_in[12];
  float* out = (float*)d_out;

  // f16 elems: 4*4096 (WT) + 512 (WsT) + 2*6144 (Q,K) + 6144 (Vt) + 6144 (X)
  //            + 8*16*96 (P) = 53760  -> 107520 B
  // f32 elems: 768 (coef) + 768 (inv) + 256 (bias) = 1792 -> 7168 B
  const size_t smem = 53760 * sizeof(_Float16) + 1792 * sizeof(float); // 114688 B

  const int BN = 4 * 170;  // B * N = 680 workgroups, one sequence each
  MultiHeadAttention_gauss_wmma_kernel<<<dim3(BN), dim3(256), smem, stream>>>(
      q, k, v, Wq, bq, Wk, bk, Wv, bv, Wo, bo, Ws, bs, out);
}
